// SingleHeadAttention_77610059039352
// MI455X (gfx1250) — matrix operand order repaired and checked by reading, not compiled, not run on hardware
//
#include <hip/hip_runtime.h>
#include <hip/hip_bf16.h>

// ---------------------------------------------------------------------------
// Problem constants (from reference): T,S=2048, B=8, E,H=1024, SCALE=H^-0.5
// ---------------------------------------------------------------------------
constexpr int T_ = 2048;
constexpr int S_ = 2048;
constexpr int B_ = 8;
constexpr int E_ = 1024;
constexpr int H_ = 1024;
constexpr float SCALE_ = 0.03125f; // 1/sqrt(1024)

typedef __attribute__((ext_vector_type(16))) __bf16 v16bf;
typedef __attribute__((ext_vector_type(8)))  float  v8f;

// ---------------------------------------------------------------------------
// WMMA fragment loaders (CDNA5 ISA 7.12.2, 16-bit operands, wave32)
//  A (16x32 MxK): lane m=lane&15; hi=lane>>4 selects K {hi*8..+7} U {16+hi*8..+7}
//  B (32x16 KxN): lane n=lane&15; hi=lane>>4 selects K {hi*8..+7} U {16+hi*8..+7}
// Sources are row-major [row, K] with row stride `ld` (elements).
// ---------------------------------------------------------------------------
template <typename T>
__device__ inline v16bf load_a_frag(const T* base, long ld, int lane, int kk) {
  const int r  = lane & 15;
  const int hi = lane >> 4;
  const T* p = base + (long)r * ld + kk + hi * 8;
  if constexpr (sizeof(T) == 4) { // f32 source: convert to bf16 on load
    v16bf o;
#pragma unroll
    for (int j = 0; j < 8; ++j) {
      o[j]     = (__bf16)p[j];
      o[j + 8] = (__bf16)p[j + 16];
    }
    return o;
  } else {                        // bf16 source: two 16B chunks
    union { v16bf v; uint4 q[2]; } f;
    f.q[0] = *(const uint4*)(p);
    f.q[1] = *(const uint4*)(p + 16);
    return f.v;
  }
}

__device__ inline v16bf load_b_frag(const __bf16* base, long ld, int lane, int kk) {
  const int n  = lane & 15;
  const int hi = lane >> 4;
  const __bf16* p = base + (long)n * ld + kk + hi * 8;
  union { v16bf v; uint4 q[2]; } f;
  f.q[0] = *(const uint4*)(p);
  f.q[1] = *(const uint4*)(p + 16);
  return f.v;
}

// ---------------------------------------------------------------------------
// Weight-norm row scale: s[r] = g[r] * rsqrt(sum_c V[r,c]^2)
// ---------------------------------------------------------------------------
__global__ void rowscale_kernel(const float* __restrict__ V,
                                const float* __restrict__ g,
                                float* __restrict__ s, int cols) {
  __shared__ float red[256];
  const int r = blockIdx.x;
  float sum = 0.f;
  for (int c = threadIdx.x; c < cols; c += blockDim.x) {
    float x = V[(size_t)r * cols + c];
    sum += x * x;
  }
  red[threadIdx.x] = sum;
  __syncthreads();
  for (int off = 128; off > 0; off >>= 1) {
    if ((int)threadIdx.x < off) red[threadIdx.x] += red[threadIdx.x + off];
    __syncthreads();
  }
  if (threadIdx.x == 0) s[r] = g[r] * rsqrtf(red[0]);
}

// W_bf16[r,c] = bf16( V[r,c] * s[r] )
__global__ void wconvert_kernel(const float* __restrict__ V,
                                const float* __restrict__ s,
                                __bf16* __restrict__ W,
                                int rows, int cols) {
  size_t n = (size_t)rows * cols;
  size_t stride = (size_t)gridDim.x * blockDim.x;
  for (size_t i = (size_t)blockIdx.x * blockDim.x + threadIdx.x; i < n; i += stride) {
    size_t r = i / (size_t)cols;
    W[i] = (__bf16)(V[i] * s[r]);
  }
}

// ---------------------------------------------------------------------------
// Generic batched WMMA GEMM:  C[z] = (A[z] @ W[z]^T + bias) * scale
//   A: [M,K] row-major (ld=lda, batch stride aBS), TA = float or __bf16
//   W: [N,K] row-major bf16 (ld=ldw, batch stride wBS)
//   C: [M,N] (ld=ldc, batch stride cBS), TC = float or __bf16
// Block: 256 thr = 8 waves; wave -> 32(M) x 64(N) (2 A-frags x 4 B-frags,
// 8 WMMAs per 32-wide k-step). grid: (N/64, M/256, nbatch)
// ---------------------------------------------------------------------------
template <typename TA, typename TC>
__global__ void gemm_wn_kernel(const TA* __restrict__ A, long lda, long aBS,
                               const __bf16* __restrict__ W, long ldw, long wBS,
                               const float* __restrict__ bias, float scale,
                               TC* __restrict__ C, long ldc, long cBS,
                               int K) {
  const int lane = threadIdx.x & 31;
  const int wave = threadIdx.x >> 5;
  const int z  = blockIdx.z;
  const int m0 = blockIdx.y * 256 + wave * 32;
  const int n0 = blockIdx.x * 64;

  const TA*     Ab = A + (size_t)z * aBS + (size_t)m0 * lda;
  const __bf16* Wb = W + (size_t)z * wBS + (size_t)n0 * ldw;

  v8f acc[2][4];
#pragma unroll
  for (int r = 0; r < 2; ++r)
#pragma unroll
    for (int j = 0; j < 4; ++j)
#pragma unroll
      for (int i = 0; i < 8; ++i) acc[r][j][i] = 0.f;

#pragma unroll 1
  for (int kk = 0; kk < K; kk += 32) {
    v16bf a0 = load_a_frag<TA>(Ab, lda, lane, kk);
    v16bf a1 = load_a_frag<TA>(Ab + (size_t)16 * lda, lda, lane, kk);
#pragma unroll
    for (int j = 0; j < 4; ++j) {
      v16bf b = load_b_frag(Wb + (size_t)(16 * j) * ldw, ldw, lane, kk);
      acc[0][j] = __builtin_amdgcn_wmma_f32_16x16x32_bf16(
          false, a0, false, b, (short)0, acc[0][j], false, false);
      acc[1][j] = __builtin_amdgcn_wmma_f32_16x16x32_bf16(
          false, a1, false, b, (short)0, acc[1][j], false, false);
    }
  }

  const int hi = lane >> 4, lo = lane & 15;
  TC* Cb = C + (size_t)z * cBS;
#pragma unroll
  for (int r = 0; r < 2; ++r) {
#pragma unroll
    for (int j = 0; j < 4; ++j) {
      const int n = n0 + 16 * j + lo;
      const float bv = bias ? bias[n] : 0.f;
#pragma unroll
      for (int i = 0; i < 8; ++i) {
        const int m = m0 + 16 * r + i + 8 * hi;
        float v = (acc[r][j][i] + bv) * scale;
        Cb[(size_t)m * ldc + n] = (TC)v;
      }
    }
  }
}

// ---------------------------------------------------------------------------
// Attention scores + online softmax stats.
//   qp,kp: projected q/k, bf16, layout [T(S),B,H] -> row stride B*H
//   P    : raw scores bf16 [B,T,S];  m,l: per-row max / sum-exp, f32 [B*T]
// grid (T/128, B), block 256; wave -> 16 query rows x 64 score cols per
// outer step (4 accumulators, Q-fragment reused 4x). Inner k-loop is kept
// rolled (#pragma unroll 1) so fragment loads stay loop-carried and cannot
// be hoisted + spilled to scratch.
// ---------------------------------------------------------------------------
__global__ void attn_scores_kernel(const __bf16* __restrict__ qp,
                                   const __bf16* __restrict__ kp,
                                   __bf16* __restrict__ P,
                                   float* __restrict__ mrow,
                                   float* __restrict__ lrow) {
  const int lane = threadIdx.x & 31;
  const int wave = threadIdx.x >> 5;
  const int b  = blockIdx.y;
  const int t0 = blockIdx.x * 128 + wave * 16;
  const long ld = (long)B_ * H_;

  const __bf16* Ab = qp + (size_t)t0 * ld + (size_t)b * H_;
  const __bf16* Kb = kp + (size_t)b * H_;

  const int hi = lane >> 4, lo = lane & 15;
  float m8[8], l8[8];
#pragma unroll
  for (int i = 0; i < 8; ++i) { m8[i] = -__builtin_inff(); l8[i] = 0.f; }

#pragma unroll 1
  for (int s0 = 0; s0 < S_; s0 += 64) {
    v8f acc[4];
#pragma unroll
    for (int j = 0; j < 4; ++j)
#pragma unroll
      for (int i = 0; i < 8; ++i) acc[j][i] = 0.f;

    const __bf16* Kt = Kb + (size_t)s0 * ld;
#pragma unroll 1
    for (int kk = 0; kk < H_; kk += 32) {
      v16bf a = load_a_frag<__bf16>(Ab, ld, lane, kk);
#pragma unroll
      for (int j = 0; j < 4; ++j) {
        v16bf bb = load_b_frag(Kt + (size_t)(16 * j) * ld, ld, lane, kk);
        acc[j] = __builtin_amdgcn_wmma_f32_16x16x32_bf16(
            false, a, false, bb, (short)0, acc[j], false, false);
      }
    }

    // online softmax stats over these 64 columns + raw score store
#pragma unroll
    for (int i = 0; i < 8; ++i) {
      float tmax = acc[0][i];
#pragma unroll
      for (int j = 1; j < 4; ++j) tmax = fmaxf(tmax, acc[j][i]);
      for (int off = 1; off < 16; off <<= 1)
        tmax = fmaxf(tmax, __shfl_xor(tmax, off, 32));
      const float mn = fmaxf(m8[i], tmax);
      float ps = 0.f;
#pragma unroll
      for (int j = 0; j < 4; ++j) ps += __expf(acc[j][i] - mn);
      for (int off = 1; off < 16; off <<= 1)
        ps += __shfl_xor(ps, off, 32);
      l8[i] = l8[i] * __expf(m8[i] - mn) + ps;
      m8[i] = mn;
      const int t = t0 + i + 8 * hi;
#pragma unroll
      for (int j = 0; j < 4; ++j)
        P[((size_t)b * T_ + t) * S_ + s0 + 16 * j + lo] = (__bf16)acc[j][i];
    }
  }
  if (lo == 0) {
#pragma unroll
    for (int i = 0; i < 8; ++i) {
      const int t = t0 + i + 8 * hi;
      mrow[(size_t)b * T_ + t] = m8[i];
      lrow[(size_t)b * T_ + t] = l8[i];
    }
  }
}

// P[i] = exp(P[i] - m[row]) / l[row]   (row = i / S)
__global__ void softmax_norm_kernel(__bf16* __restrict__ P,
                                    const float* __restrict__ m,
                                    const float* __restrict__ l) {
  size_t n = (size_t)B_ * T_ * S_;
  size_t stride = (size_t)gridDim.x * blockDim.x;
  for (size_t i = (size_t)blockIdx.x * blockDim.x + threadIdx.x; i < n; i += stride) {
    size_t row = i / (size_t)S_;
    float v = __expf((float)P[i] - m[row]) / l[row];
    P[i] = (__bf16)v;
  }
}

// vt[b][h][s] = vp[(s*B+b)*H + h]   (make V K-contiguous for the B fragment)
__global__ void vtranspose_kernel(const __bf16* __restrict__ vp,
                                  __bf16* __restrict__ vt) {
  size_t n = (size_t)B_ * H_ * S_;
  size_t stride = (size_t)gridDim.x * blockDim.x;
  for (size_t i = (size_t)blockIdx.x * blockDim.x + threadIdx.x; i < n; i += stride) {
    int s = (int)(i % S_);
    size_t rem = i / S_;
    int h = (int)(rem % H_);
    int b = (int)(rem / H_);
    vt[i] = vp[((size_t)s * B_ + b) * H_ + h];
  }
}

// ---------------------------------------------------------------------------
// Host-side orchestration
// ---------------------------------------------------------------------------
static inline size_t aln(size_t x) { return (x + 255) & ~(size_t)255; }

extern "C" void kernel_launch(void* const* d_in, const int* in_sizes, int n_in,
                              void* d_out, int out_size, void* d_ws, size_t ws_size,
                              hipStream_t stream) {
  (void)in_sizes; (void)n_in; (void)out_size; (void)ws_size;

  const float* query = (const float*)d_in[0];   // [T,B,E]
  const float* key_  = (const float*)d_in[1];   // [S,B,E]
  const float* value = (const float*)d_in[2];   // [S,B,E]
  const float* Vq = (const float*)d_in[3];      // [H,E]
  const float* Vk = (const float*)d_in[4];
  const float* Vv = (const float*)d_in[5];
  const float* Vo = (const float*)d_in[6];      // [E,H]
  const float* gq = (const float*)d_in[7];  const float* bq = (const float*)d_in[8];
  const float* gk = (const float*)d_in[9];  const float* bk = (const float*)d_in[10];
  const float* gv = (const float*)d_in[11]; const float* bv = (const float*)d_in[12];
  const float* go = (const float*)d_in[13]; const float* bo = (const float*)d_in[14];
  float* out = (float*)d_out;                   // [T,B,E]

  // ---- workspace carve-up (byte offsets, 256B aligned) ----
  char* w = (char*)d_ws;
  size_t off = 0;
  auto carve = [&](size_t bytes) { char* p = w + off; off = aln(off + bytes); return p; };

  float*  sq = (float*)carve((size_t)H_ * 4);
  float*  sk = (float*)carve((size_t)H_ * 4);
  float*  sv = (float*)carve((size_t)H_ * 4);
  float*  so = (float*)carve((size_t)E_ * 4);
  __bf16* Wq = (__bf16*)carve((size_t)H_ * E_ * 2);
  __bf16* Wk = (__bf16*)carve((size_t)H_ * E_ * 2);
  __bf16* Wv = (__bf16*)carve((size_t)H_ * E_ * 2);
  __bf16* Wo = (__bf16*)carve((size_t)E_ * H_ * 2);
  __bf16* qp = (__bf16*)carve((size_t)T_ * B_ * H_ * 2);  // [T,B,H]
  __bf16* kp = (__bf16*)carve((size_t)S_ * B_ * H_ * 2);
  __bf16* vp = (__bf16*)carve((size_t)S_ * B_ * H_ * 2);
  __bf16* P  = (__bf16*)carve((size_t)B_ * T_ * S_ * 2);  // [B,T,S]
  float*  mr = (float*)carve((size_t)B_ * T_ * 4);
  float*  lr = (float*)carve((size_t)B_ * T_ * 4);
  __bf16* vt = (__bf16*)carve((size_t)B_ * H_ * S_ * 2);  // [B,H,S]
  __bf16* at = (__bf16*)carve((size_t)T_ * B_ * H_ * 2);  // attn [T,B,H]

  const dim3 blk(256);

  // 1) weight-norm scales
  rowscale_kernel<<<dim3(H_), blk, 0, stream>>>(Vq, gq, sq, E_);
  rowscale_kernel<<<dim3(H_), blk, 0, stream>>>(Vk, gk, sk, E_);
  rowscale_kernel<<<dim3(H_), blk, 0, stream>>>(Vv, gv, sv, E_);
  rowscale_kernel<<<dim3(E_), blk, 0, stream>>>(Vo, go, so, H_);

  // 2) bf16 normalized weights
  wconvert_kernel<<<dim3(2048), blk, 0, stream>>>(Vq, sq, Wq, H_, E_);
  wconvert_kernel<<<dim3(2048), blk, 0, stream>>>(Vk, sk, Wk, H_, E_);
  wconvert_kernel<<<dim3(2048), blk, 0, stream>>>(Vv, sv, Wv, H_, E_);
  wconvert_kernel<<<dim3(2048), blk, 0, stream>>>(Vo, so, Wo, E_, H_);

  // 3) projections: [T*B,E] @ [H,E]^T -> bf16 [T*B,H]
  {
    const int MB = (T_ * B_) / 256;   // 64
    const int NB = H_ / 64;           // 16
    gemm_wn_kernel<float, __bf16><<<dim3(NB, MB, 1), blk, 0, stream>>>(
        query, E_, 0, Wq, E_, 0, bq, SCALE_, qp, H_, 0, E_);
    gemm_wn_kernel<float, __bf16><<<dim3(NB, MB, 1), blk, 0, stream>>>(
        key_, E_, 0, Wk, E_, 0, bk, 1.0f, kp, H_, 0, E_);
    gemm_wn_kernel<float, __bf16><<<dim3(NB, MB, 1), blk, 0, stream>>>(
        value, E_, 0, Wv, E_, 0, bv, 1.0f, vp, H_, 0, E_);
  }

  // 4) scores + softmax stats (QK^T, online max/sumexp)
  attn_scores_kernel<<<dim3(T_ / 128, B_), blk, 0, stream>>>(qp, kp, P, mr, lr);

  // 5) normalize probabilities in place
  softmax_norm_kernel<<<dim3(8192), blk, 0, stream>>>(P, mr, lr);

  // 6) V transpose to [B,H,S]
  vtranspose_kernel<<<dim3(8192), blk, 0, stream>>>(vp, vt);

  // 7) attn = P_b[T,S] @ V_b^T[H,S]^T -> bf16 [T,B,H]  (batched over B)
  gemm_wn_kernel<__bf16, __bf16><<<dim3(H_ / 64, T_ / 256, B_), blk, 0, stream>>>(
      P, S_, (long)T_ * S_, vt, S_, (long)H_ * S_, nullptr, 1.0f,
      at, (long)B_ * H_, (long)H_, S_);

  // 8) out = attn[T*B,H] @ Wo[E,H]^T + bo -> f32 [T,B,E]
  gemm_wn_kernel<__bf16, float><<<dim3(E_ / 64, (T_ * B_) / 256, 1), blk, 0, stream>>>(
      at, H_, 0, Wo, H_, 0, bo, 1.0f, out, E_, 0, H_);
}
